// SparseMoEBlock_824633721566
// MI455X (gfx1250) — matrix operand order repaired and checked by reading, not compiled, not run on hardware
//
#include <hip/hip_runtime.h>
#include <hip/hip_bf16.h>

#define BB   8
#define SS   4096
#define DD   1024
#define EE   8
#define II   2048
#define KKE  1024      // tokens per expert (top-k K)
#define ISHD 2048
#define BSTOK (BB*SS)

typedef __bf16 bf16;
typedef __bf16 v16bf __attribute__((ext_vector_type(16)));
typedef __bf16 v4bf  __attribute__((ext_vector_type(4)));
typedef float  v8f   __attribute__((ext_vector_type(8)));
typedef unsigned int u32x4 __attribute__((ext_vector_type(4)));
typedef int          i32x4 __attribute__((ext_vector_type(4)));
typedef int          i32x8 __attribute__((ext_vector_type(8)));

#if defined(__has_builtin)
# if __has_builtin(__builtin_amdgcn_tensor_load_to_lds)
#  define HAVE_TDM 1
# endif
#endif
#ifndef HAVE_TDM
# define HAVE_TDM 0
#endif

// LDS tiles: 128 rows x 32 K-halves, padded to 40 halves/row (80 B rows, 16B aligned)
#define LDS_STRIDE 40
#define TILE_HALVES (128 * LDS_STRIDE)

#if HAVE_TDM
// --------------------------------------------------------------- TDM 2D tile load
// Loads a 128-row x 32-element bf16 tile (row stride = stride_elems) from global
// into LDS at lds_byte_off, inserting 4 DWORDs of padding after every 16 DWORDs
// (64B data row -> 80B LDS row == LDS_STRIDE halves). Issued once per wave.
__device__ __forceinline__ void tdm_load_tile(const bf16* gsrc, unsigned lds_byte_off,
                                              unsigned stride_elems, unsigned rows_total) {
    unsigned long long ga = (unsigned long long)(uintptr_t)gsrc;
    u32x4 g0;
    g0[0] = 1u;                                            // count=1, user mode
    g0[1] = lds_byte_off;                                  // lds_addr
    g0[2] = (unsigned)ga;                                  // global_addr[95:64]
    g0[3] = (unsigned)((ga >> 32) & 0x01ffffffull) | (2u << 30);  // addr hi | type=2
    unsigned td0 = stride_elems;                           // tensor_dim0 (elements)
    unsigned td1 = rows_total;                             // tensor_dim1 (rows)
    i32x8 g1;
    g1[0] = (int)((1u << 16)      // data_size = 2 bytes
                | (1u << 20)      // pad_enable
                | (3u << 22)      // pad_interval: 16 DWORDs
                | (3u << 25));    // pad_amount:   4 DWORDs
    g1[1] = (int)((td0 & 0xffffu) << 16);                          // tensor_dim0 lo
    g1[2] = (int)(((td0 >> 16) & 0xffffu) | ((td1 & 0xffffu) << 16));
    g1[3] = (int)(((td1 >> 16) & 0xffffu) | (32u << 16));          // tile_dim0 = 32
    g1[4] = 128;                                                   // tile_dim1 = 128
    g1[5] = (int)td0;                                              // dim0 stride lo
    g1[6] = 0;
    g1[7] = 0;
    i32x4 gz; gz[0] = 0; gz[1] = 0; gz[2] = 0; gz[3] = 0;
#if defined(__clang_major__) && (__clang_major__ >= 23)
    i32x8 gz8; gz8[0]=0; gz8[1]=0; gz8[2]=0; gz8[3]=0; gz8[4]=0; gz8[5]=0; gz8[6]=0; gz8[7]=0;
    __builtin_amdgcn_tensor_load_to_lds(g0, g1, gz, gz, gz8, 0);
#else
    __builtin_amdgcn_tensor_load_to_lds(g0, g1, gz, gz, 0);
#endif
}
__device__ __forceinline__ unsigned lds_off(const void* p) {
    return (unsigned)(uintptr_t)p;   // low 32 bits of LDS-aperture generic address
}
#endif // HAVE_TDM

// ---------------------------------------------------------------- cast fp32->bf16
__global__ void cast4_kernel(const float4* __restrict__ s, v4bf* __restrict__ d, size_t n4) {
    size_t i = (size_t)blockIdx.x * blockDim.x + threadIdx.x;
    size_t stride = (size_t)gridDim.x * blockDim.x;
    for (; i < n4; i += stride) {
        float4 f = s[i];
        v4bf o;
        o[0] = (bf16)f.x; o[1] = (bf16)f.y; o[2] = (bf16)f.z; o[3] = (bf16)f.w;
        d[i] = o;
    }
}

// ---------------------------------------------------------------- gate + softmax
// one wave (32 lanes) per token; writes affinity transposed: (B, E, S)
__global__ __launch_bounds__(256) void gate_softmax_kernel(
        const float* __restrict__ x, const float* __restrict__ gw,
        float* __restrict__ aff) {
    int lane = threadIdx.x & 31;
    int wave = threadIdx.x >> 5;
    int tok  = blockIdx.x * 8 + wave;          // BSTOK/8 blocks
    int b = tok / SS, s = tok % SS;

    float acc[EE];
#pragma unroll
    for (int e = 0; e < EE; ++e) acc[e] = 0.f;

    const float* xr = x + (size_t)tok * DD;
    for (int d0 = lane; d0 < DD; d0 += 32) {
        float xv = xr[d0];
        const float* g = gw + d0 * EE;
#pragma unroll
        for (int e = 0; e < EE; ++e) acc[e] += xv * g[e];
    }
#pragma unroll
    for (int e = 0; e < EE; ++e) {
#pragma unroll
        for (int off = 16; off > 0; off >>= 1)
            acc[e] += __shfl_xor(acc[e], off, 32);
    }
    if (lane == 0) {
        float m = acc[0];
#pragma unroll
        for (int e = 1; e < EE; ++e) m = fmaxf(m, acc[e]);
        float ex[EE]; float sum = 0.f;
#pragma unroll
        for (int e = 0; e < EE; ++e) { ex[e] = __expf(acc[e] - m); sum += ex[e]; }
        float inv = 1.f / sum;
#pragma unroll
        for (int e = 0; e < EE; ++e)
            aff[((size_t)(b * EE + e)) * SS + s] = ex[e] * inv;
    }
}

// ---------------------------------------------------------------- top-K per (b,e)
__global__ __launch_bounds__(512) void topk_kernel(
        const float* __restrict__ aff, int* __restrict__ idx,
        float* __restrict__ gateval) {
    __shared__ unsigned long long keys[SS];     // 32 KB
    int be = blockIdx.x;
    const float* a = aff + (size_t)be * SS;
    for (int i = threadIdx.x; i < SS; i += blockDim.x)
        keys[i] = ((unsigned long long)__float_as_uint(a[i]) << 32) | (unsigned)i;
    __syncthreads();
    for (int k = 2; k <= SS; k <<= 1) {
        for (int j = k >> 1; j > 0; j >>= 1) {
            for (int i = threadIdx.x; i < SS; i += blockDim.x) {
                int ixj = i ^ j;
                if (ixj > i) {
                    unsigned long long u0 = keys[i], u1 = keys[ixj];
                    bool desc = ((i & k) == 0);
                    if (desc ? (u0 < u1) : (u0 > u1)) { keys[i] = u1; keys[ixj] = u0; }
                }
            }
            __syncthreads();
        }
    }
    for (int m = threadIdx.x; m < KKE; m += blockDim.x) {
        unsigned long long key = keys[m];
        idx[(size_t)be * KKE + m]     = (int)(key & 0xffffffffull);
        gateval[(size_t)be * KKE + m] = __uint_as_float((unsigned)(key >> 32));
    }
}

// ---------------------------------------------------------------- WMMA fragment helpers
__device__ __forceinline__ v16bf load_a_frag(const bf16* As, int mw, int lane) {
    union { v16bf v; uint4 q[2]; } f;
    int row = mw + (lane & 15);
    int hs  = lane >> 4;
    const uint4* p = (const uint4*)(As + row * LDS_STRIDE);
    f.q[0] = p[hs];          // lanes 0-15: K0..7 / 16-31: K8..15
    f.q[1] = p[2 + hs];      // lanes 0-15: K16..23 / 16-31: K24..31
    return f.v;
}

__device__ __forceinline__ v16bf load_b_frag(const bf16* Bs, int nw, int lane) {
    union { v16bf v; uint4 q[2]; } f;
    int row = nw + (lane & 15);
    int kh  = lane >> 4;     // lanes 0-15: K0..7,K16..23 ; 16-31: K8..15,K24..31
    const uint4* p = (const uint4*)(Bs + row * LDS_STRIDE);
    f.q[0] = p[kh];
    f.q[1] = p[2 + kh];
    return f.v;
}

// ---------------------------------------------------------------- fused gate/up GEMM
// H[g, m, n] = silu(X@Wg^T) * (X@Wu^T); A rows gathered via index (vector loads),
// Wg/Wu tiles streamed by the Tensor Data Mover, double buffered.
__global__ __launch_bounds__(256) void gateup_wmma_kernel(
        const bf16* __restrict__ X,        // token matrix (rows, Dk)
        const bf16* __restrict__ Wg,       // (wmod, Ncols, Dk)
        const bf16* __restrict__ Wu,
        const int*  __restrict__ gather,   // null => row = global row
        bf16* __restrict__ H,              // (G, Mrows, Ncols)
        int Mrows, int Ncols, int Dk, int wmod) {
    __shared__ __align__(16) bf16 As[TILE_HALVES];
    __shared__ __align__(16) bf16 Bg[2][TILE_HALVES];
    __shared__ __align__(16) bf16 Bu[2][TILE_HALVES];

    const int g   = blockIdx.z;
    const int m0  = blockIdx.y * 128;
    const int n0  = blockIdx.x * 128;
    const int tid = threadIdx.x;
    const int lane = tid & 31;
    const int mw   = (tid >> 5) * 16;

    const int r  = tid >> 1;                   // 0..127 load row
    const int c0 = (tid & 1) * 16;             // 0 or 16 halves

    int token = gather ? gather[(size_t)g * Mrows + (m0 + r)] : (m0 + r);
    const bf16* xrow = gather ? X + ((size_t)(g / EE) * SS + token) * Dk
                              : X + (size_t)token * Dk;
    const bf16* wgtile = Wg + (size_t)(g % wmod) * Ncols * Dk + (size_t)n0 * Dk;
    const bf16* wutile = Wu + (size_t)(g % wmod) * Ncols * Dk + (size_t)n0 * Dk;
#if !HAVE_TDM
    const bf16* wgrow = wgtile + (size_t)r * Dk;
    const bf16* wurow = wutile + (size_t)r * Dk;
#endif

    v8f accG[8], accU[8];
#pragma unroll
    for (int t = 0; t < 8; ++t)
#pragma unroll
        for (int j = 0; j < 8; ++j) { accG[t][j] = 0.f; accU[t][j] = 0.f; }

#if HAVE_TDM
    if (tid < 32) {                            // pre-issue first chunk
        tdm_load_tile(wgtile, lds_off(&Bg[0][0]), (unsigned)Dk, (unsigned)Ncols);
        tdm_load_tile(wutile, lds_off(&Bu[0][0]), (unsigned)Dk, (unsigned)Ncols);
    }
#endif

    int it = 0;
    for (int d0 = 0; d0 < Dk; d0 += 32, ++it) {
        const int cur = it & 1;
        __syncthreads();                       // prior consumers done; buffers free
        // ---- A tile (gathered) : global -> LDS via VGPRs
        {
            const uint4* pa = (const uint4*)(xrow + d0 + c0);
            uint4 a0 = pa[0], a1 = pa[1];
            *(uint4*)(As + r * LDS_STRIDE + c0)     = a0;
            *(uint4*)(As + r * LDS_STRIDE + c0 + 8) = a1;
            if (d0 + 32 < Dk) __builtin_prefetch(xrow + d0 + 32 + c0, 0, 3);
        }
#if HAVE_TDM
        if (tid < 32) {
            if (d0 + 32 < Dk) {                // issue next chunk into other buffer
                tdm_load_tile(wgtile + (d0 + 32), lds_off(&Bg[cur ^ 1][0]),
                              (unsigned)Dk, (unsigned)Ncols);
                tdm_load_tile(wutile + (d0 + 32), lds_off(&Bu[cur ^ 1][0]),
                              (unsigned)Dk, (unsigned)Ncols);
                __builtin_amdgcn_s_wait_tensorcnt(2);  // in-order => current landed
            } else {
                __builtin_amdgcn_s_wait_tensorcnt(0);
            }
        }
#else
        {
            const uint4* pg = (const uint4*)(wgrow + d0 + c0);
            const uint4* pu = (const uint4*)(wurow + d0 + c0);
            uint4 g0 = pg[0], g1 = pg[1];
            uint4 u0 = pu[0], u1 = pu[1];
            *(uint4*)(&Bg[cur][0] + r * LDS_STRIDE + c0)     = g0;
            *(uint4*)(&Bg[cur][0] + r * LDS_STRIDE + c0 + 8) = g1;
            *(uint4*)(&Bu[cur][0] + r * LDS_STRIDE + c0)     = u0;
            *(uint4*)(&Bu[cur][0] + r * LDS_STRIDE + c0 + 8) = u1;
        }
#endif
        __syncthreads();                       // A stores + TDM completion visible

        v16bf fa = load_a_frag(As, mw, lane);
#pragma unroll
        for (int nt = 0; nt < 8; ++nt) {
            v16bf fg = load_b_frag(&Bg[cur][0], nt * 16, lane);
            v16bf fu = load_b_frag(&Bu[cur][0], nt * 16, lane);
            accG[nt] = __builtin_amdgcn_wmma_f32_16x16x32_bf16(
                false, fa, false, fg, (short)0, accG[nt], false, false);
            accU[nt] = __builtin_amdgcn_wmma_f32_16x16x32_bf16(
                false, fa, false, fu, (short)0, accU[nt], false, false);
        }
    }

    const int hs = lane >> 4, nl = lane & 15;
#pragma unroll
    for (int nt = 0; nt < 8; ++nt) {
#pragma unroll
        for (int rr = 0; rr < 8; ++rr) {
            float gv = accG[nt][rr];
            float uv = accU[nt][rr];
            float hv = gv * uv / (1.f + __expf(-gv));   // silu(g)*u
            int lm = mw + rr + 8 * hs;
            size_t orow = (size_t)g * Mrows + (m0 + lm);
            H[orow * Ncols + (n0 + nt * 16 + nl)] = (bf16)hv;
        }
    }
}

// ---------------------------------------------------------------- down-proj GEMM
// Both operands are regular row-major tiles -> full TDM double buffering.
// gather!=null: expert mode -> scale by gating, atomic scatter-add into OUT.
// gather==null: shared mode -> direct store (initializes OUT).
__global__ __launch_bounds__(256) void down_wmma_kernel(
        const bf16* __restrict__ H,        // (G*Mrows, Dk)
        const bf16* __restrict__ Wd,       // (wmod, Ncols, Dk)
        const int*   __restrict__ gather,
        const float* __restrict__ gateval,
        float* __restrict__ OUT,           // (B*S, D)
        int Mrows, int Ncols, int Dk, int wmod) {
    __shared__ __align__(16) bf16 As[2][TILE_HALVES];
    __shared__ __align__(16) bf16 Bd[2][TILE_HALVES];

    const int g   = blockIdx.z;
    const int m0  = blockIdx.y * 128;
    const int n0  = blockIdx.x * 128;
    const int tid = threadIdx.x;
    const int lane = tid & 31;
    const int mw   = (tid >> 5) * 16;

    const bf16* atile = H  + ((size_t)g * Mrows + m0) * Dk;
    const bf16* btile = Wd + (size_t)(g % wmod) * Ncols * Dk + (size_t)n0 * Dk;
#if !HAVE_TDM
    const int r  = tid >> 1;
    const int c0 = (tid & 1) * 16;
    const bf16* hrow = atile + (size_t)r * Dk;
    const bf16* wrow = btile + (size_t)r * Dk;
#endif

    v8f acc[8];
#pragma unroll
    for (int t = 0; t < 8; ++t)
#pragma unroll
        for (int j = 0; j < 8; ++j) acc[t][j] = 0.f;

#if HAVE_TDM
    if (tid < 32) {
        tdm_load_tile(atile, lds_off(&As[0][0]), (unsigned)Dk, (unsigned)Mrows);
        tdm_load_tile(btile, lds_off(&Bd[0][0]), (unsigned)Dk, (unsigned)Ncols);
    }
#endif

    int it = 0;
    for (int d0 = 0; d0 < Dk; d0 += 32, ++it) {
        const int cur = it & 1;
        __syncthreads();
#if HAVE_TDM
        if (tid < 32) {
            if (d0 + 32 < Dk) {
                tdm_load_tile(atile + (d0 + 32), lds_off(&As[cur ^ 1][0]),
                              (unsigned)Dk, (unsigned)Mrows);
                tdm_load_tile(btile + (d0 + 32), lds_off(&Bd[cur ^ 1][0]),
                              (unsigned)Dk, (unsigned)Ncols);
                __builtin_amdgcn_s_wait_tensorcnt(2);
            } else {
                __builtin_amdgcn_s_wait_tensorcnt(0);
            }
        }
#else
        {
            const uint4* pa = (const uint4*)(hrow + d0 + c0);
            const uint4* pb = (const uint4*)(wrow + d0 + c0);
            uint4 a0 = pa[0], a1 = pa[1];
            uint4 b0 = pb[0], b1 = pb[1];
            *(uint4*)(&As[cur][0] + r * LDS_STRIDE + c0)     = a0;
            *(uint4*)(&As[cur][0] + r * LDS_STRIDE + c0 + 8) = a1;
            *(uint4*)(&Bd[cur][0] + r * LDS_STRIDE + c0)     = b0;
            *(uint4*)(&Bd[cur][0] + r * LDS_STRIDE + c0 + 8) = b1;
        }
#endif
        __syncthreads();

        v16bf fa = load_a_frag(&As[cur][0], mw, lane);
#pragma unroll
        for (int nt = 0; nt < 8; ++nt) {
            v16bf fb = load_b_frag(&Bd[cur][0], nt * 16, lane);
            acc[nt] = __builtin_amdgcn_wmma_f32_16x16x32_bf16(
                false, fa, false, fb, (short)0, acc[nt], false, false);
        }
    }

    const int hs = lane >> 4, nl = lane & 15;
#pragma unroll
    for (int nt = 0; nt < 8; ++nt) {
#pragma unroll
        for (int rr = 0; rr < 8; ++rr) {
            float v = acc[nt][rr];
            int lm  = mw + rr + 8 * hs;
            int col = n0 + nt * 16 + nl;
            if (gather) {
                int mg     = m0 + lm;
                int token  = gather[(size_t)g * Mrows + mg];
                float sc   = gateval[(size_t)g * Mrows + mg];
                atomicAdd(&OUT[((size_t)(g / EE) * SS + token) * Ncols + col], v * sc);
            } else {
                OUT[(size_t)(m0 + lm) * Ncols + col] = v;
            }
        }
    }
}

// ---------------------------------------------------------------- launcher
extern "C" void kernel_launch(void* const* d_in, const int* in_sizes, int n_in,
                              void* d_out, int out_size, void* d_ws, size_t ws_size,
                              hipStream_t stream) {
    (void)in_sizes; (void)n_in; (void)out_size; (void)ws_size;
    const float* x   = (const float*)d_in[0];
    const float* gw  = (const float*)d_in[1];
    const float* egw = (const float*)d_in[2];
    const float* euw = (const float*)d_in[3];
    const float* edw = (const float*)d_in[4];
    const float* sgw = (const float*)d_in[5];
    const float* suw = (const float*)d_in[6];
    const float* sdw = (const float*)d_in[7];
    float* out = (float*)d_out;

    char* ws = (char*)d_ws;
    size_t off = 0;
    float* aff      = (float*)(ws + off); off += (size_t)BB * EE * SS * 4;
    int*   tk_idx   = (int*)  (ws + off); off += (size_t)BB * EE * KKE * 4;
    float* tk_gate  = (float*)(ws + off); off += (size_t)BB * EE * KKE * 4;
    bf16*  x_bf     = (bf16*) (ws + off); off += (size_t)BSTOK * DD * 2;
    bf16*  wg_bf    = (bf16*) (ws + off); off += (size_t)EE * II * DD * 2;
    bf16*  wu_bf    = (bf16*) (ws + off); off += (size_t)EE * II * DD * 2;
    bf16*  wd_bf    = (bf16*) (ws + off); off += (size_t)EE * DD * II * 2;
    bf16*  sg_bf    = (bf16*) (ws + off); off += (size_t)ISHD * DD * 2;
    bf16*  su_bf    = (bf16*) (ws + off); off += (size_t)ISHD * DD * 2;
    bf16*  sd_bf    = (bf16*) (ws + off); off += (size_t)DD * ISHD * 2;
    bf16*  h_exp    = (bf16*) (ws + off); off += (size_t)BB * EE * KKE * II * 2;
    bf16*  h_sh     = (bf16*) (ws + off); off += (size_t)BSTOK * ISHD * 2;

    // 1) casts fp32 -> bf16
    auto cast = [&](const float* s, bf16* d, size_t n) {
        cast4_kernel<<<4096, 256, 0, stream>>>((const float4*)s, (v4bf*)d, n / 4);
    };
    cast(x,   x_bf,  (size_t)BSTOK * DD);
    cast(egw, wg_bf, (size_t)EE * II * DD);
    cast(euw, wu_bf, (size_t)EE * II * DD);
    cast(edw, wd_bf, (size_t)EE * DD * II);
    cast(sgw, sg_bf, (size_t)ISHD * DD);
    cast(suw, su_bf, (size_t)ISHD * DD);
    cast(sdw, sd_bf, (size_t)DD * ISHD);

    // 2) gating softmax -> (B,E,S)
    gate_softmax_kernel<<<BSTOK / 8, 256, 0, stream>>>(x, gw, aff);

    // 3) top-K per (b,e)
    topk_kernel<<<BB * EE, 512, 0, stream>>>(aff, tk_idx, tk_gate);

    // 4) expert fused gate/up: (B*E groups) K x I, inner D
    gateup_wmma_kernel<<<dim3(II / 128, KKE / 128, BB * EE), 256, 0, stream>>>(
        x_bf, wg_bf, wu_bf, tk_idx, h_exp, KKE, II, DD, EE);

    // 5) shared fused gate/up: all tokens, B*S x ISH, inner D
    gateup_wmma_kernel<<<dim3(ISHD / 128, BSTOK / 128, 1), 256, 0, stream>>>(
        x_bf, sg_bf, su_bf, nullptr, h_sh, BSTOK, ISHD, DD, 1);

    // 6) shared down-proj: direct store, initializes OUT
    down_wmma_kernel<<<dim3(DD / 128, BSTOK / 128, 1), 256, 0, stream>>>(
        h_sh, sd_bf, nullptr, nullptr, out, BSTOK, DD, ISHD, 1);

    // 7) expert down-proj: gated atomic scatter-add into OUT
    down_wmma_kernel<<<dim3(DD / 128, KKE / 128, BB * EE), 256, 0, stream>>>(
        h_exp, wd_bf, tk_idx, tk_gate, out, KKE, DD, II, EE);
}
